// GATHead_19628000542915
// MI455X (gfx1250) — compile-verified, matrix operand order checked by reading, not run on hardware
//
#include <hip/hip_runtime.h>
#include <hip/hip_bf16.h>

typedef __attribute__((ext_vector_type(16))) _Float16 v16h;
typedef __attribute__((ext_vector_type(8)))  _Float16 v8h;
typedef __attribute__((ext_vector_type(4)))  _Float16 v4h;
typedef __attribute__((ext_vector_type(8)))  float    v8f;

constexpr int  BB   = 8;
constexpr int  NN   = 2048;
constexpr int  FIN  = 128;
constexpr int  FOUT = 64;
constexpr float NEGV = -1000000000.0f;

// Padded K strides (keep 16-B alignment, spread WMMA fragment loads over banks)
constexpr int KP1 = FIN + 8;   // 136 halves = 272 B rows for h / W^T tiles
constexpr int KP2 = 40;        // 40 halves  =  80 B rows for P / whJ^T tiles

// A/B 16-bit fragment: lane half-group `hi` covers K = hi*8+{0..7} and
// hi*8+16+{0..7} -> two contiguous 16-byte runs => 2x ds_load_b128.
__device__ __forceinline__ v16h ld_frag(const _Float16* base) {
    v8h lo = *(const v8h*)(base);
    v8h hh = *(const v8h*)(base + 16);
    return __builtin_shufflevector(lo, hh, 0, 1, 2, 3, 4, 5, 6, 7,
                                           8, 9, 10, 11, 12, 13, 14, 15);
}

// ---------------------------------------------------------------------------
// Kernel 1: wh = h @ W  (f16 WMMA, f32 accum), plus wh1 = wh@a0, wh2 = wh@a1
// grid = B * N/64 blocks of 128 threads (4 waves); each wave does a 16x64 tile
// ---------------------------------------------------------------------------
__global__ void __launch_bounds__(128)
gat_wh_kernel(const float* __restrict__ h, const float* __restrict__ W,
              const float* __restrict__ a, _Float16* __restrict__ wh,
              float* __restrict__ wh1, float* __restrict__ wh2)
{
    __shared__ _Float16 sWT[FOUT][KP1];    // W transposed: [col][k], K innermost
    __shared__ _Float16 sH[64][KP1];       // 64 rows of h, K innermost
    __shared__ float    sWh[4][16][FOUT + 1];

    const int tid  = threadIdx.x;
    const int wave = tid >> 5;
    const int lane = tid & 31;
    const int b    = blockIdx.x / (NN / 64);
    const int i0   = (blockIdx.x % (NN / 64)) * 64;
    const int iw   = i0 + wave * 16;

    // load W (row-major [k][f]) -> sWT[f][k] in f16; float4 global loads
    {
        const float4* W4 = (const float4*)W;
        for (int q = tid; q < FIN * FOUT / 4; q += 128) {
            float4 d = W4[q];
            const int k  = (q * 4) / FOUT;
            const int f0 = (q * 4) % FOUT;
            sWT[f0 + 0][k] = (_Float16)d.x;
            sWT[f0 + 1][k] = (_Float16)d.y;
            sWT[f0 + 2][k] = (_Float16)d.z;
            sWT[f0 + 3][k] = (_Float16)d.w;
        }
    }
    // load this wave's 16 h rows -> sH (f16); float4 loads, b64 LDS stores
    {
        const float4* h4 = (const float4*)(h + ((size_t)b * NN + iw) * FIN);
        for (int q = lane; q < 16 * FIN / 4; q += 32) {
            float4 d = h4[q];
            const int r  = (q * 4) / FIN;
            const int k0 = (q * 4) % FIN;
            v4h x = { (_Float16)d.x, (_Float16)d.y, (_Float16)d.z, (_Float16)d.w };
            *(v4h*)&sH[wave * 16 + r][k0] = x;
        }
    }
    __syncthreads();

    const int rowm = lane & 15;
    const int hi   = lane >> 4;

    v8f c[4] = {};
    #pragma unroll
    for (int ks = 0; ks < 4; ++ks) {              // K = 128 in 4 steps of 32
        const int kb = ks * 32 + hi * 8;
        const v16h af = ld_frag(&sH[wave * 16 + rowm][kb]);
        #pragma unroll
        for (int nf = 0; nf < 4; ++nf) {          // 4 x 16 output columns
            const v16h bf = ld_frag(&sWT[nf * 16 + rowm][kb]);
            c[nf] = __builtin_amdgcn_wmma_f32_16x16x32_f16(
                false, af, false, bf, (short)0, c[nf], false, false);
        }
    }

    // stage C fragments (lane holds col = lane%16, vgpr v -> row v + 8*hi)
    #pragma unroll
    for (int nf = 0; nf < 4; ++nf)
        #pragma unroll
        for (int v = 0; v < 8; ++v)
            sWh[wave][v + 8 * hi][nf * 16 + rowm] = c[nf][v];
    __syncthreads();

    // write wh (f16) to workspace
    {
        _Float16* whp = wh + ((size_t)b * NN + iw) * FOUT;
        for (int idx = lane; idx < 16 * FOUT; idx += 32)
            whp[idx] = (_Float16)sWh[wave][idx / FOUT][idx % FOUT];
    }
    // per-row projections wh1/wh2
    if (lane < 16) {
        float s1 = 0.f, s2 = 0.f;
        #pragma unroll 8
        for (int f = 0; f < FOUT; ++f) {
            const float v = sWh[wave][lane][f];
            s1 += v * a[f];
            s2 += v * a[FOUT + f];
        }
        wh1[(size_t)b * NN + iw + lane] = s1;
        wh2[(size_t)b * NN + iw + lane] = s2;
    }
}

// ---------------------------------------------------------------------------
// Kernel 2: flash-style masked softmax + attention @ wh, single pass over adj
// grid = B * N/64 blocks of 128 threads; wave w owns rows iw..iw+15
// ---------------------------------------------------------------------------
__global__ void __launch_bounds__(128)
gat_attn_kernel(const float* __restrict__ adj, const _Float16* __restrict__ wh,
                const float* __restrict__ wh1, const float* __restrict__ wh2,
                float* __restrict__ out)
{
    __shared__ _Float16 sWhJT[FOUT][KP2]; // wh tile transposed: [col][j-K]
    __shared__ float    sWh2J[32];
    __shared__ float    sE[4][16][33];    // per-wave score tile (odd stride)
    __shared__ _Float16 sP[4][16][KP2];   // per-wave prob tile (f16, K innermost)
    __shared__ float    sScale[4][16];
    __shared__ float    sSum[4][16];
    __shared__ float    sWh1[4][16];

    const int tid  = threadIdx.x;
    const int wave = tid >> 5;
    const int lane = tid & 31;
    const int b    = blockIdx.x / (NN / 64);
    const int i0   = (blockIdx.x % (NN / 64)) * 64;
    const int iw   = i0 + wave * 16;

    if (lane < 16) sWh1[wave][lane] = wh1[(size_t)b * NN + iw + lane];

    float m_r = -INFINITY;   // running row max (lanes 0-15 own row=lane)
    float s_r = 0.f;         // running row sum
    v8f c[4] = {};           // 16x64 f32 accumulator in WMMA C layout

    const float* adjB = adj + (size_t)b * NN * NN;
    __syncthreads();

    for (int jt = 0; jt < NN / 32; ++jt) {
        const int j0 = jt * 32;

        // load 32x64 wh tile (transposed into LDS) + wh2 slice
        {
            const uint4* src4 = (const uint4*)(wh + ((size_t)b * NN + j0) * FOUT);
            for (int q = tid; q < 32 * FOUT / 8; q += 128) {   // 8 halves/load
                uint4 d = src4[q];
                const _Float16* hp = (const _Float16*)&d;
                const int j  = (q * 8) / FOUT;
                const int f0 = (q * 8) % FOUT;
                #pragma unroll
                for (int u = 0; u < 8; ++u) sWhJT[f0 + u][j] = hp[u];
            }
            if (tid < 32) sWh2J[tid] = wh2[(size_t)b * NN + j0 + tid];
        }
        // prefetch next adj tile rows (global_prefetch_b8)
        if (jt + 1 < NN / 32)
            __builtin_prefetch(&adjB[(size_t)(iw + (lane & 15)) * NN + j0 + 32], 0, 0);
        __syncthreads();

        // scores: lane handles column j0+lane across the wave's 16 rows
        {
            const float w2 = sWh2J[lane];
            #pragma unroll 4
            for (int r = 0; r < 16; ++r) {
                float e = sWh1[wave][r] + w2;
                e = (e > 0.f) ? e : 0.2f * e;                     // leaky relu
                const float am = adjB[(size_t)(iw + r) * NN + j0 + lane];
                e = (am == NEGV) ? NEGV : e;                      // mask
                sE[wave][r][lane] = e;
            }
        }
        __syncthreads();

        // online softmax update: lanes 0-15, row = lane
        if (lane < 16) {
            const int r = lane;
            float rowmax = sE[wave][r][0];
            #pragma unroll
            for (int k = 1; k < 32; ++k) rowmax = fmaxf(rowmax, sE[wave][r][k]);
            const float mnew  = fmaxf(m_r, rowmax);
            const float scale = __expf(m_r - mnew);
            float ps = 0.f;
            #pragma unroll
            for (int k = 0; k < 32; ++k) {
                const float p = __expf(sE[wave][r][k] - mnew);
                sP[wave][r][k] = (_Float16)p;
                ps += p;
            }
            s_r = s_r * scale + ps;
            m_r = mnew;
            sScale[wave][r] = scale;
        }
        __syncthreads();

        // rescale accumulator and accumulate P(16x32) @ whJ(32x64) via WMMA
        {
            const int rowm = lane & 15;
            const int hi   = lane >> 4;
            #pragma unroll
            for (int nf = 0; nf < 4; ++nf)
                #pragma unroll
                for (int v = 0; v < 8; ++v)
                    c[nf][v] *= sScale[wave][v + 8 * hi];

            const v16h af = ld_frag(&sP[wave][rowm][hi * 8]);
            #pragma unroll
            for (int nf = 0; nf < 4; ++nf) {
                const v16h bf = ld_frag(&sWhJT[nf * 16 + rowm][hi * 8]);
                c[nf] = __builtin_amdgcn_wmma_f32_16x16x32_f16(
                    false, af, false, bf, (short)0, c[nf], false, false);
            }
        }
        __syncthreads();   // protect sWhJT/sE/sP before next iteration
    }

    // finalize: divide by softmax sum, ELU, write out
    if (lane < 16) sSum[wave][lane] = s_r;
    __syncthreads();
    {
        const int rowm = lane & 15;
        const int hi   = lane >> 4;
        #pragma unroll
        for (int nf = 0; nf < 4; ++nf)
            #pragma unroll
            for (int v = 0; v < 8; ++v) {
                const int row = v + 8 * hi;
                float x = c[nf][v] / sSum[wave][row];
                x = (x > 0.f) ? x : (__expf(x) - 1.f);            // ELU
                out[((size_t)b * NN + iw + row) * FOUT + nf * 16 + rowm] = x;
            }
    }
}

extern "C" void kernel_launch(void* const* d_in, const int* in_sizes, int n_in,
                              void* d_out, int out_size, void* d_ws, size_t ws_size,
                              hipStream_t stream) {
    (void)in_sizes; (void)n_in; (void)out_size; (void)ws_size;
    const float* h    = (const float*)d_in[0];
    const float* adj  = (const float*)d_in[1];
    const float* W    = (const float*)d_in[2];
    const float* a    = (const float*)d_in[3];
    float*       outp = (float*)d_out;

    // workspace layout: wh f16 [B*N*FOUT] | wh1 f32 [B*N] | wh2 f32 [B*N]
    _Float16* wh  = (_Float16*)d_ws;
    float*    wh1 = (float*)((char*)d_ws + (size_t)BB * NN * FOUT * sizeof(_Float16));
    float*    wh2 = wh1 + (size_t)BB * NN;

    dim3 grid(BB * (NN / 64)), block(128);
    gat_wh_kernel<<<grid, block, 0, stream>>>(h, W, a, wh, wh1, wh2);
    gat_attn_kernel<<<grid, block, 0, stream>>>(adj, wh, wh1, wh2, outp);
}
